// BSplineBasis_58265526337935
// MI455X (gfx1250) — matrix operand order repaired and checked by reading, not compiled, not run on hardware
//
#include <hip/hip_runtime.h>
#include <cstdint>
#include <cstddef>

// ---------------------------------------------------------------------------
// KAN layer: spline_out(scale-folded) + residual, restructured as ONE bf16
// WMMA GEMM:  out[4096,1024] = A[4096,12288] * W^T[12288,1024] + bias
//   A[b, i*12+m] = bspline_basis_m(clamp(x[b,i]))   (m = 0..10)
//   A[b, i*12+11] = x[b,i]                          (raw x, residual path)
//   W[o, i*12+m] = spline_coeffs[o,i,m]*scale[o];  W[o,i*12+11] = res_w[o,i]
// Workspace: [scale: 4KB][W bf16: 25.2MB][A bf16: 100.7MB]  (~126 MB total)
//
// GEMM: block = 8 waves laid out 2(M) x 4(N) -> 64 rows x 512 cols.
// A slabs (64 rows x 64 K, 8KB) are staged global->LDS with
// GLOBAL_LOAD_ASYNC_TO_LDS_B128 (ASYNCcnt), 4-slab ring, prefetch depth 3;
// waves read A fragments via ds_load_b128 (conflict-free 144B row stride).
// B (W) streams through WGP$/L2 with plain global_load_b128.
// ---------------------------------------------------------------------------

#define BATCH   4096
#define IN_F    1024
#define OUT_F   1024
#define KPI     12                    // 11 basis values + raw x per input feat
#define K_DIM   (IN_F * KPI)          // 12288
#define GRID_MIN (-1.0f)
#define H_INV    4.0f                 // 1 / h, h = 0.25

// GEMM staging geometry
#define CH        64                  // K elements per staged slab
#define NCH       (K_DIM / CH)        // 192 slabs
#define ROWS      64                  // A rows per block
#define ROWSTRIDE 72                  // ushorts per LDS row (144 B, padded)
#define SLABELEMS (ROWS * ROWSTRIDE)  // 4608 ushorts = 9216 B
#define NBUF      4                   // slab ring depth

typedef __bf16 bf16_t;
typedef bf16_t v16bf __attribute__((ext_vector_type(16)));
typedef float  v8f   __attribute__((ext_vector_type(8)));

union FragBF16 { uint4 u[2]; v16bf v; };

__device__ __forceinline__ unsigned short f2bf(float f) {
    union { float f; unsigned int u; } c; c.f = f;
    unsigned int u = c.u;
    u += 0x7FFFu + ((u >> 16) & 1u);      // round-to-nearest-even
    return (unsigned short)(u >> 16);
}

// ------------------------- scale[o] = mean(scale_base[o,:]) -----------------
__global__ __launch_bounds__(256) void kan_scale_kernel(const float* __restrict__ sb,
                                                        float* __restrict__ scale) {
    __shared__ float red[256];
    const int o = blockIdx.x;
    float acc = 0.f;
    for (int i = threadIdx.x; i < IN_F; i += 256) acc += sb[(size_t)o * IN_F + i];
    red[threadIdx.x] = acc;
    __syncthreads();
    for (int w = 128; w > 0; w >>= 1) {
        if (threadIdx.x < w) red[threadIdx.x] += red[threadIdx.x + w];
        __syncthreads();
    }
    if (threadIdx.x == 0) scale[o] = red[0] * (1.0f / IN_F);
}

// ------------------- build W bf16 [OUT_F][K_DIM] (scale folded) -------------
__global__ __launch_bounds__(256) void kan_build_w_kernel(const float* __restrict__ coeffs,
                                                          const float* __restrict__ resw,
                                                          const float* __restrict__ scale,
                                                          unsigned short* __restrict__ W) {
    const int idx = blockIdx.x * 256 + threadIdx.x;     // o*IN_F + i
    const int o = idx / IN_F;
    const int i = idx - o * IN_F;
    const float s = scale[o];
    const float* c = coeffs + (size_t)idx * 11;
    union { unsigned short s16[12]; uint2 v2[3]; } row;
#pragma unroll
    for (int m = 0; m < 11; ++m) row.s16[m] = f2bf(c[m] * s);
    row.s16[11] = f2bf(resw[idx]);
    uint2* dst = ((uint2*)W) + (size_t)o * (K_DIM / 4) + (size_t)i * 3;  // 24B rows
    dst[0] = row.v2[0]; dst[1] = row.v2[1]; dst[2] = row.v2[2];
}

// ------------- build A bf16 [BATCH][K_DIM]: uniform cubic B-spline ----------
__global__ __launch_bounds__(256) void kan_build_a_kernel(const float* __restrict__ X,
                                                          unsigned short* __restrict__ A) {
    const int idx = blockIdx.x * 256 + threadIdx.x;     // b*IN_F + i
    const int b = idx / IN_F;
    const int i = idx - b * IN_F;
    const float x  = X[idx];
    const float xc = fminf(fmaxf(x, -1.0f), 1.0f);
    const float u  = (xc - GRID_MIN) * H_INV;           // in [0, 8]
    int c = (int)floorf(u);
    c = (c < 0) ? 0 : ((c > 8) ? 8 : c);
    const float t   = u - (float)c;                     // local coord in [0,1)
    const float omt = 1.0f - t;
    const float t2 = t * t, t3 = t2 * t;
    float w[4];
    w[0] = omt * omt * omt * (1.0f / 6.0f);
    w[1] = (3.0f * t3 - 6.0f * t2 + 4.0f) * (1.0f / 6.0f);
    w[2] = (-3.0f * t3 + 3.0f * t2 + 3.0f * t + 1.0f) * (1.0f / 6.0f);
    w[3] = t3 * (1.0f / 6.0f);                          // == 0 when c == 8
    union { unsigned short s16[12]; uint2 v2[3]; } row;
#pragma unroll
    for (int m = 0; m < 12; ++m) row.s16[m] = 0;
#pragma unroll
    for (int r = 0; r < 4; ++r) {
        const int m = c + r;                            // nonzero bases c..c+3
        if (m < 11) row.s16[m] = f2bf(w[r]);
    }
    row.s16[11] = f2bf(x);                              // raw x (residual path)
    uint2* dst = ((uint2*)A) + (size_t)b * (K_DIM / 4) + (size_t)i * 3;
    dst[0] = row.v2[0]; dst[1] = row.v2[1]; dst[2] = row.v2[2];
}

// --------------------------- WMMA bf16 GEMM ---------------------------------
// Block: 256 threads = 8 waves laid out 2 (M-groups) x 4 (N-groups).
// Wave (mg,ng): rows [blockY*64 + mg*32, +32) (2 m-tiles),
// cols [blockX*512 + ng*128, +128) (8 n-tiles).
// A slab pipeline: ring of 4 LDS slabs, async prefetch distance 3.
__global__ __launch_bounds__(256) void kan_gemm_kernel(const unsigned short* __restrict__ A,
                                                       const unsigned short* __restrict__ W,
                                                       const float* __restrict__ bias,
                                                       float* __restrict__ out) {
    __shared__ __align__(16) unsigned short ldsA[NBUF * SLABELEMS];   // 36 KB

    const int tid   = threadIdx.x;
    const int wave  = tid >> 5;
    const int lane  = tid & 31;
    const int lhalf = lane >> 4;        // 0: lanes 0-15, 1: lanes 16-31
    const int l16   = lane & 15;
    const int mg    = wave >> 2;        // 0..1
    const int ng    = wave & 3;         // 0..3

    const int row0 = blockIdx.y * 64;
    const int col0 = blockIdx.x * 512 + ng * 128;

    // --- async-copy assignment: thread copies 2x16B of each 8KB slab -------
    // slab layout: 64 rows x 64 K bf16, LDS row stride 144 B (bank-safe).
    const int cpRow = tid >> 2;                         // 0..63
    const int cpC   = tid & 3;                          // 16B chunk 0..3 (+4 via offset:64)
    const unsigned short* aSrc = A + (size_t)(row0 + cpRow) * K_DIM + cpC * 8;
    const unsigned ldsBase = (unsigned)(uintptr_t)&ldsA[0];   // LDS byte offset
    const unsigned cpDst   = ldsBase + (unsigned)(cpRow * 144 + cpC * 16);

    // --- A fragment source in LDS (ISA 16-bit A layout) --------------------
    // lane reads row mg*32 + mt*16 + l16; K chunk kbase = lhalf*8:
    // u[0] = K kbase..+7 (byte lhalf*16), u[1] = K kbase+16..+23 (+32 B).
    const unsigned short* fragPtr = &ldsA[(mg * 32 + l16) * ROWSTRIDE + lhalf * 8];

    // --- B fragment base (ISA 16-bit B layout): col N = l16,
    // lane group 0 -> K 0..7,16..23, group 1 -> K 8..15,24..31.
    const unsigned short* bBase = W + (size_t)(col0 + l16) * K_DIM + lhalf * 8;

    v8f acc[2][8];
#pragma unroll
    for (int mt = 0; mt < 2; ++mt)
#pragma unroll
        for (int nt = 0; nt < 8; ++nt) acc[mt][nt] = (v8f){0.f,0.f,0.f,0.f,0.f,0.f,0.f,0.f};

    // --- prologue: prefetch slabs 0..2 (2 async b128 per thread per slab) --
#pragma unroll
    for (int p = 0; p < 3; ++p) {
        const unsigned short* s = aSrc + (size_t)p * CH;
        const unsigned d = cpDst + (unsigned)(p % NBUF) * (SLABELEMS * 2);
        asm volatile("global_load_async_to_lds_b128 %0, %1, off"
                     :: "v"(d), "v"(s) : "memory");
        asm volatile("global_load_async_to_lds_b128 %0, %1, off offset:64"
                     :: "v"(d), "v"(s) : "memory");
    }

    for (int ch = 0; ch < NCH; ++ch) {
        // oldest slab (ch) complete: 6 outstanding -> wait to <= 4.
        asm volatile("s_wait_asynccnt 0x4" ::: "memory");
        __syncthreads();    // slab ch visible to all; ring slot (ch+3)%4 free

        // prefetch slab ch+3 (tail: re-fetch last slab into unread slots)
        {
            int cpf = ch + 3; if (cpf > NCH - 1) cpf = NCH - 1;
            const unsigned short* s = aSrc + (size_t)cpf * CH;
            const unsigned d = cpDst + (unsigned)((ch + 3) % NBUF) * (SLABELEMS * 2);
            asm volatile("global_load_async_to_lds_b128 %0, %1, off"
                         :: "v"(d), "v"(s) : "memory");
            asm volatile("global_load_async_to_lds_b128 %0, %1, off offset:64"
                         :: "v"(d), "v"(s) : "memory");
        }

        const unsigned bufOff = (unsigned)(ch % NBUF) * SLABELEMS;   // elements
#pragma unroll
        for (int s = 0; s < 2; ++s) {                   // two K=32 sub-steps
            FragBF16 a[2];
#pragma unroll
            for (int mt = 0; mt < 2; ++mt) {
                const unsigned short* pa = fragPtr + bufOff + mt * 16 * ROWSTRIDE + s * 32;
                a[mt].u[0] = *(const uint4*)(pa);       // ds_load_b128
                a[mt].u[1] = *(const uint4*)(pa + 16);  // ds_load_b128
            }
            const size_t kOff = (size_t)ch * CH + (size_t)s * 32;
#pragma unroll
            for (int nt = 0; nt < 8; ++nt) {
                FragBF16 bfr;
                const unsigned short* pb = bBase + (size_t)nt * 16 * K_DIM + kOff;
                bfr.u[0] = *(const uint4*)(pb);
                bfr.u[1] = *(const uint4*)(pb + 16);
                acc[0][nt] = __builtin_amdgcn_wmma_f32_16x16x32_bf16(
                    false, a[0].v, false, bfr.v, (short)0, acc[0][nt], false, false);
                acc[1][nt] = __builtin_amdgcn_wmma_f32_16x16x32_bf16(
                    false, a[1].v, false, bfr.v, (short)0, acc[1][nt], false, false);
            }
        }
    }

    // C/D layout: VGPR v holds M = v + 8*lhalf, N = lane%16.
#pragma unroll
    for (int nt = 0; nt < 8; ++nt) {
        const int col = col0 + nt * 16 + l16;
        const float bv = bias[col];
#pragma unroll
        for (int mt = 0; mt < 2; ++mt) {
#pragma unroll
            for (int v = 0; v < 8; ++v) {
                const int row = row0 + mg * 32 + mt * 16 + lhalf * 8 + v;
                out[(size_t)row * OUT_F + col] = acc[mt][nt][v] + bv;
            }
        }
    }
}

// ---------------------------------------------------------------------------
extern "C" void kernel_launch(void* const* d_in, const int* in_sizes, int n_in,
                              void* d_out, int out_size, void* d_ws, size_t ws_size,
                              hipStream_t stream) {
    (void)in_sizes; (void)n_in; (void)out_size; (void)ws_size;
    const float* x      = (const float*)d_in[0];   // [4096,1024]
    const float* coeffs = (const float*)d_in[1];   // [1024,1024,11]
    const float* resw   = (const float*)d_in[2];   // [1024,1024]
    const float* bias   = (const float*)d_in[3];   // [1024]
    const float* sbase  = (const float*)d_in[4];   // [1024,1024]
    float* out = (float*)d_out;                    // [4096,1024]

    char* ws = (char*)d_ws;
    float*          scale = (float*)ws;                                       // 4 KB
    unsigned short* Wbf   = (unsigned short*)(ws + 4096);                     // 25.2 MB
    unsigned short* Abf   = (unsigned short*)(ws + 4096 + (size_t)OUT_F * K_DIM * 2);

    kan_scale_kernel<<<OUT_F, 256, 0, stream>>>(sbase, scale);
    kan_build_w_kernel<<<(OUT_F * IN_F) / 256, 256, 0, stream>>>(coeffs, resw, scale, Wbf);
    kan_build_a_kernel<<<(BATCH * IN_F) / 256, 256, 0, stream>>>(x, Abf);
    kan_gemm_kernel<<<dim3(OUT_F / 512, BATCH / 64), 256, 0, stream>>>(Abf, Wbf, bias, out);
}
